// ForkAttention_83700322664593
// MI455X (gfx1250) — compile-verified, matrix operand order checked by reading, not run on hardware
//
#include <hip/hip_runtime.h>
#include <hip/hip_bf16.h>
#include <math.h>

// ForkAttention on MI455X (gfx1250): 5 bf16-WMMA GEMMs + masked softmax.
// All GEMMs consume B column-major (BT row-major [N][K]) so A/B tile staging
// and fragment loads share one code path; only x needs an explicit transpose.
// 256x128 block tile, 8 wave32s, 64x64 per wave (16 WMMAs / K-step / wave),
// LDS rows padded to 40 halfs (80B) for bank-conflict-free ds_load_b128.

typedef __attribute__((ext_vector_type(16))) __bf16 v16bf;
typedef __attribute__((ext_vector_type(8)))  __bf16 v8bf;
typedef __attribute__((ext_vector_type(8)))  float  v8f;

#define B_DIM 4
#define C_DIM 2048
#define E_DIM 1024

#define BM 256
#define BN 128
#define BK 32
#define LDS_STRIDE 40   // halfs per LDS row (32 data + 8 pad), 80 bytes

__device__ __forceinline__ unsigned short f2bf(float f) {
  union { float f; unsigned int u; } cv; cv.f = f;
  unsigned int u = cv.u;
  u += 0x7FFFu + ((u >> 16) & 1u);   // round-to-nearest-even
  return (unsigned short)(u >> 16);
}

// ---------------- conversion kernels ----------------

__global__ void cvt_bf16(const float* __restrict__ s, unsigned short* __restrict__ d, int n) {
  int i = blockIdx.x * 256 + threadIdx.x;
  if (i < n) d[i] = f2bf(s[i]);
}

// x[b][c][e] (f32) -> x_bf (same layout) and xT_bf[b][e][c]
__global__ void cvt_x(const float* __restrict__ x,
                      unsigned short* __restrict__ xb,
                      unsigned short* __restrict__ xtb) {
  long long i = (long long)blockIdx.x * 256 + threadIdx.x;
  int e = (int)(i % E_DIM);
  int c = (int)((i / E_DIM) % C_DIM);
  int b = (int)(i / ((long long)E_DIM * C_DIM));
  unsigned short v = f2bf(x[i]);
  xb[i] = v;
  xtb[((long long)b * E_DIM + e) * C_DIM + c] = v;
}

// ---------------- tiled bf16 WMMA GEMM ----------------
// D[b][M][N] = A[b][M][K] * Bt[b][N][K]^T (+ bias).
// Wave grid 4(M) x 2(N); each wave: 4x4 accumulators of 16x16.
// BIAS_MODE: 0 none, 1 per-N, 2 per-M.  OUT_BF16: bf16 vs f32 output.

template <int BIAS_MODE, bool OUT_BF16>
__global__ void __launch_bounds__(256) gemm_wmma_bf16(
    const __bf16* __restrict__ A,  unsigned long long strideA, int lda,
    const __bf16* __restrict__ Bt, unsigned long long strideB, int ldb,
    void* __restrict__ Dv,         unsigned long long strideD, int ldd,
    const float* __restrict__ bias, int Ktot)
{
  __shared__ __attribute__((aligned(16))) __bf16 sA[BM * LDS_STRIDE];  // 20 KB
  __shared__ __attribute__((aligned(16))) __bf16 sB[BN * LDS_STRIDE];  // 10 KB

  const int tid   = threadIdx.x;
  const int lane  = tid & 31;
  const int wv    = tid >> 5;
  const int waveM = wv & 3;        // 4 waves along M (64 rows each)
  const int waveN = wv >> 2;       // 2 waves along N (64 cols each)
  const int blockM = blockIdx.y * BM;
  const int blockN = blockIdx.x * BN;

  const __bf16* Ab = A  + (unsigned long long)blockIdx.z * strideA;
  const __bf16* Bb = Bt + (unsigned long long)blockIdx.z * strideB;

  v8f acc[4][4];
  for (int i = 0; i < 4; ++i)
    for (int j = 0; j < 4; ++j)
      for (int r = 0; r < 8; ++r) acc[i][j][r] = 0.0f;

  // cooperative tile load: 2 threads per row, 16 halfs (32B) each;
  // A covers 256 rows in 2 passes of 128, B covers 128 rows in 1 pass.
  const int ldr = tid >> 1;            // 0..127
  const int ldc = (tid & 1) * 16;      // 0 or 16 (half elements)
  const __bf16* gA0 = Ab + (unsigned long long)(blockM + ldr)       * lda + ldc;
  const __bf16* gA1 = Ab + (unsigned long long)(blockM + 128 + ldr) * lda + ldc;
  const __bf16* gB  = Bb + (unsigned long long)(blockN + ldr)       * ldb + ldc;

  for (int kb = 0; kb < Ktot; kb += BK) {
    const uint4* pA0 = (const uint4*)(gA0 + kb);
    const uint4* pA1 = (const uint4*)(gA1 + kb);
    const uint4* pB  = (const uint4*)(gB  + kb);
    uint4 a00 = pA0[0], a01 = pA0[1];
    uint4 a10 = pA1[0], a11 = pA1[1];
    uint4 b0  = pB[0],  b1  = pB[1];
    if (kb + BK < Ktot) {                       // prefetch next K-tile
      __builtin_prefetch(gA0 + kb + BK, 0, 3);
      __builtin_prefetch(gA1 + kb + BK, 0, 3);
      __builtin_prefetch(gB  + kb + BK, 0, 3);
    }
    *(uint4*)(&sA[ldr * LDS_STRIDE + ldc])           = a00;
    *(uint4*)(&sA[ldr * LDS_STRIDE + ldc + 8])       = a01;
    *(uint4*)(&sA[(128 + ldr) * LDS_STRIDE + ldc])   = a10;
    *(uint4*)(&sA[(128 + ldr) * LDS_STRIDE + ldc + 8]) = a11;
    *(uint4*)(&sB[ldr * LDS_STRIDE + ldc])           = b0;
    *(uint4*)(&sB[ldr * LDS_STRIDE + ldc + 8])       = b1;
    __syncthreads();

    // fragment loads matching ISA 16-bit 16x32 striping:
    // lanes 0-15 hold K 0..7 / 16..23, lanes 16-31 hold K 8..15 / 24..31
    const int k0 = (lane >> 4) * 8;
    const int ml = lane & 15;
    v16bf bfrag[4];
    for (int in = 0; in < 4; ++in) {
      int row = waveN * 64 + in * 16 + ml;
      v8bf lo = *(const v8bf*)(&sB[row * LDS_STRIDE + k0]);
      v8bf hi = *(const v8bf*)(&sB[row * LDS_STRIDE + 16 + k0]);
      bfrag[in] = __builtin_shufflevector(lo, hi, 0,1,2,3,4,5,6,7,8,9,10,11,12,13,14,15);
    }
    for (int im = 0; im < 4; ++im) {
      int row = waveM * 64 + im * 16 + ml;
      v8bf lo = *(const v8bf*)(&sA[row * LDS_STRIDE + k0]);
      v8bf hi = *(const v8bf*)(&sA[row * LDS_STRIDE + 16 + k0]);
      v16bf afrag = __builtin_shufflevector(lo, hi, 0,1,2,3,4,5,6,7,8,9,10,11,12,13,14,15);
      for (int in = 0; in < 4; ++in)
        acc[im][in] = __builtin_amdgcn_wmma_f32_16x16x32_bf16(
            false, afrag, false, bfrag[in], (short)0, acc[im][in], false, false);
    }
    __syncthreads();
  }

  // epilogue: C/D layout -> lane holds N=lane&15, VGPR r -> M = r + 8*(lane>=16)
  const int nloc = lane & 15;
  const int mhi  = (lane >> 4) * 8;
  for (int im = 0; im < 4; ++im) {
    for (int in = 0; in < 4; ++in) {
      int mbase = blockM + waveM * 64 + im * 16 + mhi;
      int n     = blockN + waveN * 64 + in * 16 + nloc;
      float bn = (BIAS_MODE == 1) ? bias[n] : 0.0f;
      for (int r = 0; r < 8; ++r) {
        int m = mbase + r;
        float v = acc[im][in][r];
        if (BIAS_MODE == 1) v += bn;
        else if (BIAS_MODE == 2) v += bias[m];
        if (OUT_BF16) {
          unsigned short* Dp = (unsigned short*)Dv + (unsigned long long)blockIdx.z * strideD;
          Dp[(unsigned long long)m * ldd + n] = f2bf(v);
        } else {
          float* Dp = (float*)Dv + (unsigned long long)blockIdx.z * strideD;
          Dp[(unsigned long long)m * ldd + n] = v;
        }
      }
    }
  }
}

// ---------------- masked softmax over last dim (C=2048), f32 -> bf16 ----------------

__global__ void __launch_bounds__(256) softmax_rows(
    const float* __restrict__ attn, const unsigned char* __restrict__ mask,
    unsigned short* __restrict__ out)
{
  __shared__ float red[256];
  const int t = threadIdx.x;
  const long long row = (long long)blockIdx.y * C_DIM + blockIdx.x;
  const float* rp = attn + row * C_DIM;
  const unsigned char* mp = mask + row * C_DIM;
  const float scale = 0.022097086912079612f;   // 1/sqrt(2048)

  float vals[8];
  float mx = -INFINITY;
  for (int i = 0; i < 8; ++i) {
    int j = t + i * 256;
    float v = mp[j] ? -INFINITY : rp[j] * scale;
    vals[i] = v;
    mx = fmaxf(mx, v);
  }
  red[t] = mx; __syncthreads();
  for (int s = 128; s > 0; s >>= 1) { if (t < s) red[t] = fmaxf(red[t], red[t + s]); __syncthreads(); }
  const float rowmax = red[0];
  __syncthreads();

  float sum = 0.0f;
  for (int i = 0; i < 8; ++i) { vals[i] = __expf(vals[i] - rowmax); sum += vals[i]; }
  red[t] = sum; __syncthreads();
  for (int s = 128; s > 0; s >>= 1) { if (t < s) red[t] += red[t + s]; __syncthreads(); }
  const float inv = 1.0f / red[0];

  unsigned short* op = out + row * C_DIM;
  for (int i = 0; i < 8; ++i) op[t + i * 256] = f2bf(vals[i] * inv);
}

// ---------------- launch ----------------

extern "C" void kernel_launch(void* const* d_in, const int* in_sizes, int n_in,
                              void* d_out, int out_size, void* d_ws, size_t ws_size,
                              hipStream_t stream) {
  const float* x     = (const float*)d_in[0];
  const float* fc_w  = (const float*)d_in[1];
  const float* fc_b  = (const float*)d_in[2];
  const float* alt_w = (const float*)d_in[3];
  const float* alt_b = (const float*)d_in[4];
  const float* v_w   = (const float*)d_in[5];
  const float* v_b   = (const float*)d_in[6];
  const unsigned char* mask = (const unsigned char*)d_in[7];

  char* ws = (char*)d_ws;
  size_t off = 0;
  auto take = [&](size_t bytes) -> char* {
    char* p = ws + off;
    off += bytes;
    off = (off + 255) & ~(size_t)255;
    return p;
  };

  const size_t BCE = (size_t)B_DIM * C_DIM * E_DIM;   // 8M elems
  const size_t BCC = (size_t)B_DIM * C_DIM * C_DIM;   // 16M elems

  unsigned short* x_bf    = (unsigned short*)take(BCE * 2);                    // x bf16 [B][C][E]
  unsigned short* xT_bf   = (unsigned short*)take(BCE * 2);                    // xT bf16 [B][E][C]
  unsigned short* fcw_bf  = (unsigned short*)take((size_t)E_DIM * E_DIM * 2);  // fc_w bf16 [E][E]
  unsigned short* altw_bf = (unsigned short*)take((size_t)C_DIM * C_DIM * 2);  // alt_w bf16 [C][C]
  unsigned short* vw_bf   = (unsigned short*)take((size_t)E_DIM * E_DIM * 2);  // v_w bf16 [E][E]
  unsigned short* y_bf    = (unsigned short*)take(BCE * 2);                    // y [B][C][E]
  unsigned short* yxT_bf  = (unsigned short*)take(BCE * 2);                    // yx^T [B][C(k)][E]
  unsigned short* vxT_bf  = (unsigned short*)take(BCE * 2);                    // vx^T [B][E][C(k)]
  float*          attn_f  = (float*)take(BCC * 4);                             // attn f32 [B][C][C]
  unsigned short* attn_bf = (unsigned short*)take(BCC * 2);                    // softmax(attn) bf16

  // precision conversion (+ x transpose)
  cvt_x<<<dim3((unsigned)(BCE / 256)), 256, 0, stream>>>(x, x_bf, xT_bf);
  cvt_bf16<<<dim3((E_DIM * E_DIM) / 256), 256, 0, stream>>>(fc_w, fcw_bf, E_DIM * E_DIM);
  cvt_bf16<<<dim3((C_DIM * C_DIM) / 256), 256, 0, stream>>>(alt_w, altw_bf, C_DIM * C_DIM);
  cvt_bf16<<<dim3((E_DIM * E_DIM) / 256), 256, 0, stream>>>(v_w, vw_bf, E_DIM * E_DIM);

  // G1: y[b] = x[b] @ fc_w^T + fc_b   (BT = fc_w row-major directly)
  gemm_wmma_bf16<1, true><<<dim3(E_DIM / BN, C_DIM / BM, B_DIM), 256, 0, stream>>>(
      (const __bf16*)x_bf,   (unsigned long long)C_DIM * E_DIM, E_DIM,
      (const __bf16*)fcw_bf, 0ULL,                              E_DIM,
      y_bf,                  (unsigned long long)C_DIM * E_DIM, E_DIM,
      fc_b, E_DIM);

  // G2': yxT[b][k][e] = sum_c alt_w[k][c] * x[b][c][e] + alt_b[k]  (BT = xT_bf[b])
  gemm_wmma_bf16<2, true><<<dim3(E_DIM / BN, C_DIM / BM, B_DIM), 256, 0, stream>>>(
      (const __bf16*)altw_bf, 0ULL,                              C_DIM,
      (const __bf16*)xT_bf,   (unsigned long long)E_DIM * C_DIM, C_DIM,
      yxT_bf,                 (unsigned long long)C_DIM * E_DIM, E_DIM,
      alt_b, C_DIM);

  // G4': vxT[b][e][k] = sum_e' v_w[e][e'] * x[b][k][e'] + v_b[e]  (BT = x_bf[b])
  gemm_wmma_bf16<2, true><<<dim3(C_DIM / BN, E_DIM / BM, B_DIM), 256, 0, stream>>>(
      (const __bf16*)vw_bf, 0ULL,                              E_DIM,
      (const __bf16*)x_bf,  (unsigned long long)C_DIM * E_DIM, E_DIM,
      vxT_bf,               (unsigned long long)E_DIM * C_DIM, C_DIM,
      v_b, E_DIM);

  // G3: attn[b] = y[b] @ yx[b]  (BT = yxT_bf[b]), f32 out
  gemm_wmma_bf16<0, false><<<dim3(C_DIM / BN, C_DIM / BM, B_DIM), 256, 0, stream>>>(
      (const __bf16*)y_bf,   (unsigned long long)C_DIM * E_DIM, E_DIM,
      (const __bf16*)yxT_bf, (unsigned long long)C_DIM * E_DIM, E_DIM,
      attn_f,                (unsigned long long)C_DIM * C_DIM, C_DIM,
      nullptr, E_DIM);

  // masked scaled softmax -> bf16
  softmax_rows<<<dim3(C_DIM, B_DIM), 256, 0, stream>>>(attn_f, mask, attn_bf);

  // G5: out[b] = attn[b] @ vx[b]  (BT = vxT_bf[b]), f32 -> d_out
  gemm_wmma_bf16<0, false><<<dim3(E_DIM / BN, C_DIM / BM, B_DIM), 256, 0, stream>>>(
      (const __bf16*)attn_bf, (unsigned long long)C_DIM * C_DIM, C_DIM,
      (const __bf16*)vxT_bf,  (unsigned long long)E_DIM * C_DIM, C_DIM,
      d_out,                  (unsigned long long)C_DIM * E_DIM, E_DIM,
      nullptr, C_DIM);

  (void)in_sizes; (void)n_in; (void)out_size; (void)ws_size;
}
